// GAT_NC_2499670966369
// MI455X (gfx1250) — matrix operand order repaired and checked by reading, not compiled, not run on hardware
//
#include <hip/hip_runtime.h>
#include <hip/hip_bf16.h>
#include <math.h>

// ---------------- constants (match reference) ----------------
#define NN   50000
#define EE   800000
#define INDIM 256
#define DD   64
#define HH   4
#define OUTC 40
#define SLOPE 0.2f
#define KK   256          // all GEMMs have K = 256
#define KSTEPS (KK / 32)  // 8

typedef __bf16 bf16;
typedef __attribute__((ext_vector_type(16))) __bf16 v16bf;
typedef __attribute__((ext_vector_type(8)))  float  v8f;

// ---------------- helpers ----------------
__device__ __forceinline__ bf16 f32_to_bf16_rne(float f) {
    unsigned u = __float_as_uint(f);
    unsigned r = (u + 0x7FFFu + ((u >> 16) & 1u)) >> 16;
    unsigned short s = (unsigned short)r;
    bf16 b;
    __builtin_memcpy(&b, &s, 2);
    return b;
}

// order-preserving float atomic max via int punning
__device__ __forceinline__ void atomicMaxF32(float* addr, float v) {
    int iv = __float_as_int(v);
    if (iv >= 0) atomicMax((int*)addr, iv);
    else         atomicMin((unsigned int*)addr, (unsigned int)iv);
}

__device__ __forceinline__ float lrelu(float x) { return x > 0.f ? x : SLOPE * x; }

// ---------------- elementwise kernels ----------------
__global__ void cvt_f32_bf16_k(const float* __restrict__ x, bf16* __restrict__ y, long n) {
    long i = (long)blockIdx.x * blockDim.x + threadIdx.x;
    if (i < n) y[i] = f32_to_bf16_rne(x[i]);
}

__global__ void zero_f32_k(float* __restrict__ y, long n) {
    long i = (long)blockIdx.x * blockDim.x + threadIdx.x;
    if (i < n) y[i] = 0.f;
}

__global__ void copy_f32_k(const float* __restrict__ x, float* __restrict__ y, long n) {
    long i = (long)blockIdx.x * blockDim.x + threadIdx.x;
    if (i < n) y[i] = x[i];
}

__global__ void elu_k(const float* __restrict__ x, float* __restrict__ y, long n) {
    long i = (long)blockIdx.x * blockDim.x + threadIdx.x;
    if (i < n) { float v = x[i]; y[i] = v > 0.f ? v : (__expf(v) - 1.f); }
}

__global__ void init_ms_k(float* __restrict__ mx, float* __restrict__ sm, long n) {
    long i = (long)blockIdx.x * blockDim.x + threadIdx.x;
    if (i < n) { mx[i] = -__builtin_inff(); sm[i] = 0.f; }
}

// ---------------- weight pack: row-major f32 [K x Nc] -> bf16 WMMA-B fragment order ----
// P[((kb*Nc + n)*2 + hi)*16 + t] = W[(kb*32 + hi*16 + t)*Nc + n]
__global__ void pack_w_k(const float* __restrict__ W, bf16* __restrict__ P, int K, int Nc) {
    int total = K * Nc;
    int p = blockIdx.x * blockDim.x + threadIdx.x;
    if (p >= total) return;
    int t    = p & 15;
    int hi   = (p >> 4) & 1;
    int rest = p >> 5;
    int n    = rest % Nc;
    int kb   = rest / Nc;
    int k    = kb * 32 + hi * 16 + t;
    P[p] = f32_to_bf16_rne(W[(size_t)k * Nc + n]);
}

// ---------------- bf16 WMMA GEMM: C[MxNc] = A[Mx256] * B[256xNc], f32 accumulate ------
// blockIdx.y + g0 = 64-column group (B panel staged to LDS once per block),
// blockIdx.x = strip of 8 m-tiles (one 16-row tile per wave).
// JT = number of valid 16-col tiles in the group (compile-time: 4 or 2).
// LDS layout: "unit" = 16 bf16 (32B), unit u = (kb*JT*16 + nn)*2 + hi,
// padded 16B every 4 units to break the 64B lane stride (bank conflicts).
// LBOFF is affine in (kb, j) for fixed lane: LBOFF(u0 + 32j + 2*JC*kb)
//   = LBOFF(u0) + 576*j + 36*JC*kb   (elements), so all fragment addresses
// fold into one per-lane base + compile-time ds_load immediate offsets.
#define LBOFF(u) ((u) * 16 + ((u) >> 2) * 8)

template <int JT>
__global__ void gemm_bf16_wmma_k(const bf16* __restrict__ A, const bf16* __restrict__ Bp,
                                 float* __restrict__ C, int M, int Nc, int g0) {
    constexpr int JC = JT * 16;                 // columns in this group
    constexpr int NU = KSTEPS * JC * 2;         // 32B units in the panel
    constexpr int KBSTRIDE = 36 * JC;           // LDS element stride per kb
    constexpr int JSTRIDE  = 576;               // LDS element stride per j-tile
    __shared__ bf16 lb[NU * 16 + (NU / 4) * 8];

    const int g = blockIdx.y + g0;

    // ---- cooperative stage of B panel (all kb, JC cols) into LDS ----
    for (int u = threadIdx.x; u < NU; u += blockDim.x) {
        int hi = u & 1;
        int nn = (u >> 1) % JC;
        int kb = u / (JC * 2);
        const bf16* gsrc = Bp + (((size_t)kb * Nc + g * 64 + nn) * 2 + hi) * 16;
        float4 z0 = *(const float4*)gsrc;
        float4 z1 = *(const float4*)(gsrc + 8);
        bf16* ld = &lb[LBOFF(u)];
        *(float4*)ld       = z0;                // ds_store_b128
        *(float4*)(ld + 8) = z1;
    }
    __syncthreads();

    const int wib = threadIdx.x >> 5;
    const int mt  = blockIdx.x * 8 + wib;
    if (mt >= (M >> 4)) return;                 // wave-uniform, after the barrier
    const int lane = threadIdx.x & 31;
    const int lm = lane & 15;
    const int hi = lane >> 4;
    const int m0 = mt << 4;

    v8f acc[JT] = {};
    const bf16* arow  = A + (size_t)(m0 + lm) * KK + hi * 16;
    const bf16* lbase = &lb[LBOFF(lm * 2 + hi)];   // per-lane fragment base

    #pragma unroll
    for (int kb = 0; kb < KSTEPS; ++kb) {
        union { v16bf v; float4 f[2]; } ua;
        ua.f[0] = *(const float4*)(arow + kb * 32);        // K = hi*16 + 0..7
        ua.f[1] = *(const float4*)(arow + kb * 32 + 8);    // K = hi*16 + 8..15
        if (kb + 1 < KSTEPS) __builtin_prefetch(arow + kb * 32 + 32, 0, 1); // global_prefetch_b8
        #pragma unroll
        for (int j = 0; j < JT; ++j) {
            union { v16bf v; float4 f[2]; } ub;
            const bf16* ls = lbase + kb * KBSTRIDE + j * JSTRIDE;  // immediate ds offsets
            ub.f[0] = *(const float4*)ls;        // ds_load_b128
            ub.f[1] = *(const float4*)(ls + 8);
            acc[j] = __builtin_amdgcn_wmma_f32_16x16x32_bf16(
                         false, ua.v, false, ub.v, (short)0, acc[j], false, false);
        }
    }
    float* cbase = C + (size_t)(m0 + hi * 8) * Nc + g * 64 + lm;
    #pragma unroll
    for (int j = 0; j < JT; ++j) {
        float* crow = cbase + j * 16;
        #pragma unroll
        for (int r = 0; r < 8; ++r) crow[(size_t)r * Nc] = acc[j][r];
    }
}

// ---------------- attention kernels ----------------
// wave per (node, head); lanes span d; butterfly reduce
__global__ void att_lr_k(const float* __restrict__ feat, const float* __restrict__ al,
                         const float* __restrict__ ar, float* __restrict__ el,
                         float* __restrict__ er, int n, int heads, int dh) {
    long wid = ((long)blockIdx.x * blockDim.x + threadIdx.x) >> 5;
    int lane = threadIdx.x & 31;
    if (wid >= (long)n * heads) return;
    int node = (int)(wid / heads), h = (int)(wid % heads);
    const float* f = feat + (size_t)node * heads * dh + (size_t)h * dh;
    const float* a = al + (size_t)h * dh;
    const float* b = ar + (size_t)h * dh;
    float sl = 0.f, sr = 0.f;
    for (int d = lane; d < dh; d += 32) { float v = f[d]; sl += v * a[d]; sr += v * b[d]; }
    #pragma unroll
    for (int o = 16; o > 0; o >>= 1) { sl += __shfl_xor(sl, o, 32); sr += __shfl_xor(sr, o, 32); }
    if (lane == 0) { el[wid] = sl; er[wid] = sr; }
}

__global__ void edge_max_k(const int* __restrict__ src, const int* __restrict__ dst,
                           const float* __restrict__ el, const float* __restrict__ er,
                           float* __restrict__ mx, int heads) {
    long i = (long)blockIdx.x * blockDim.x + threadIdx.x;
    if (i >= (long)EE * heads) return;
    int e = (int)(i / heads), h = (int)(i % heads);
    float x = lrelu(el[(size_t)src[e] * heads + h] + er[(size_t)dst[e] * heads + h]);
    atomicMaxF32(&mx[(size_t)dst[e] * heads + h], x);
}

__global__ void edge_sum_k(const int* __restrict__ src, const int* __restrict__ dst,
                           const float* __restrict__ el, const float* __restrict__ er,
                           const float* __restrict__ mx, float* __restrict__ sm, int heads) {
    long i = (long)blockIdx.x * blockDim.x + threadIdx.x;
    if (i >= (long)EE * heads) return;
    int e = (int)(i / heads), h = (int)(i % heads);
    size_t di = (size_t)dst[e] * heads + h;
    float x = lrelu(el[(size_t)src[e] * heads + h] + er[di]);
    atomicAdd(&sm[di], __expf(x - mx[di]));
}

// wave per (edge, head); lanes span d -> coalesced gather + coalesced f32 atomics
__global__ void edge_scatter_k(const int* __restrict__ src, const int* __restrict__ dst,
                               const float* __restrict__ el, const float* __restrict__ er,
                               const float* __restrict__ mx, const float* __restrict__ sm,
                               const float* __restrict__ feat, float* __restrict__ rst,
                               int heads, int dh) {
    long wid = ((long)blockIdx.x * blockDim.x + threadIdx.x) >> 5;
    int lane = threadIdx.x & 31;
    if (wid >= (long)EE * heads) return;
    int e = (int)(wid / heads), h = (int)(wid % heads);
    int s = src[e], d = dst[e];
    size_t di = (size_t)d * heads + h;
    float x = lrelu(el[(size_t)s * heads + h] + er[di]);
    float alpha = __expf(x - mx[di]) / sm[di];
    const float* fs = feat + (size_t)s * heads * dh + (size_t)h * dh;
    float*       rd = rst  + (size_t)d * heads * dh + (size_t)h * dh;
    for (int k = lane; k < dh; k += 32) atomicAdd(&rd[k], alpha * fs[k]);
}

__global__ void mean_heads_k(const float* __restrict__ rst, float* __restrict__ out,
                             int n, int heads, int outc) {
    long i = (long)blockIdx.x * blockDim.x + threadIdx.x;
    if (i >= (long)n * outc) return;
    int node = (int)(i / outc), c = (int)(i % outc);
    float s = 0.f;
    for (int h = 0; h < heads; ++h) s += rst[((size_t)node * heads + h) * outc + c];
    out[i] = s * (1.f / heads);
}

// ---------------- launch ----------------
static inline int gblk(long n) { return (int)((n + 255) / 256); }
static inline size_t alup(size_t x) { return (x + 255) & ~(size_t)255; }

extern "C" void kernel_launch(void* const* d_in, const int* in_sizes, int n_in,
                              void* d_out, int out_size, void* d_ws, size_t ws_size,
                              hipStream_t stream) {
    (void)in_sizes; (void)n_in; (void)out_size; (void)ws_size;
    const float* inputs = (const float*)d_in[0];
    const int*   src    = (const int*)d_in[1];
    const int*   dst    = (const int*)d_in[2];
    const float* W0     = (const float*)d_in[3];
    const float* al0    = (const float*)d_in[4];
    const float* ar0    = (const float*)d_in[5];
    const float* W1     = (const float*)d_in[6];
    const float* al1    = (const float*)d_in[7];
    const float* ar1    = (const float*)d_in[8];
    const float* Wo     = (const float*)d_in[9];
    const float* alo    = (const float*)d_in[10];
    const float* aro    = (const float*)d_in[11];
    const float* resWo  = (const float*)d_in[12];
    const float* fcW    = (const float*)d_in[13];

    // output regions (f32): logits | h | seq_fts
    float* LOGITS = (float*)d_out;
    float* HOUT   = LOGITS + (size_t)NN * OUTC;
    float* SEQ    = HOUT   + (size_t)NN * (HH * DD);

    // workspace carve-out
    char* ws = (char*)d_ws;
    size_t off = 0;
    float* FEAT = (float*)(ws + off); off = alup(off + (size_t)NN * 256 * 4);
    float* RST  = (float*)(ws + off); off = alup(off + (size_t)NN * 256 * 4);
    float* H0   = (float*)(ws + off); off = alup(off + (size_t)NN * 256 * 4);
    bf16*  ABF  = (bf16*)(ws + off);  off = alup(off + (size_t)NN * 256 * 2);
    bf16*  W0p  = (bf16*)(ws + off);  off = alup(off + (size_t)256 * 256 * 2);
    bf16*  W1p  = (bf16*)(ws + off);  off = alup(off + (size_t)256 * 256 * 2);
    bf16*  Wop  = (bf16*)(ws + off);  off = alup(off + (size_t)256 * 160 * 2);
    bf16*  rWop = (bf16*)(ws + off);  off = alup(off + (size_t)256 * 160 * 2);
    bf16*  fcp  = (bf16*)(ws + off);  off = alup(off + (size_t)256 * 64 * 2);
    float* EL   = (float*)(ws + off); off = alup(off + (size_t)NN * HH * 4);
    float* ER   = (float*)(ws + off); off = alup(off + (size_t)NN * HH * 4);
    float* MX   = (float*)(ws + off); off = alup(off + (size_t)NN * HH * 4);
    float* SM   = (float*)(ws + off); off = alup(off + (size_t)NN * HH * 4);

    const long nFeat = (long)NN * 256;
    const long nNH   = (long)NN * HH;
    const long nEH   = (long)EE * HH;
    const int  mstrips = (NN / 16 + 7) / 8;    // 391 blocks of 8 m-tiles

    // ---- pack weights to WMMA fragment order (bf16) ----
    pack_w_k<<<gblk(256 * 256), 256, 0, stream>>>(W0,    W0p,  256, 256);
    pack_w_k<<<gblk(256 * 256), 256, 0, stream>>>(W1,    W1p,  256, 256);
    pack_w_k<<<gblk(256 * 160), 256, 0, stream>>>(Wo,    Wop,  256, 160);
    pack_w_k<<<gblk(256 * 160), 256, 0, stream>>>(resWo, rWop, 256, 160);
    pack_w_k<<<gblk(256 * 64),  256, 0, stream>>>(fcW,   fcp,  256, 64);

    // GEMM launch helpers: full 64-col groups with JT=4; 32-col tail with JT=2
    auto gemm256 = [&](const bf16* a, const bf16* b, float* c) {
        gemm_bf16_wmma_k<4><<<dim3(mstrips, 4), 256, 0, stream>>>(a, b, c, NN, 256, 0);
    };
    auto gemm160 = [&](const bf16* a, const bf16* b, float* c) {
        gemm_bf16_wmma_k<4><<<dim3(mstrips, 2), 256, 0, stream>>>(a, b, c, NN, 160, 0);
        gemm_bf16_wmma_k<2><<<dim3(mstrips, 1), 256, 0, stream>>>(a, b, c, NN, 160, 2);
    };
    auto gemm64 = [&](const bf16* a, const bf16* b, float* c) {
        gemm_bf16_wmma_k<4><<<dim3(mstrips, 1), 256, 0, stream>>>(a, b, c, NN, 64, 0);
    };

    // ---- inputs -> bf16 ; seq_fts = inputs @ fc_W ----
    cvt_f32_bf16_k<<<gblk(nFeat), 256, 0, stream>>>(inputs, ABF, nFeat);
    gemm64(ABF, fcp, SEQ);

    // ================= layer 0 =================
    gemm256(ABF, W0p, FEAT);
    att_lr_k<<<gblk(nNH * 32), 256, 0, stream>>>(FEAT, al0, ar0, EL, ER, NN, HH, DD);
    init_ms_k<<<gblk(nNH), 256, 0, stream>>>(MX, SM, nNH);
    zero_f32_k<<<gblk(nFeat), 256, 0, stream>>>(RST, nFeat);
    edge_max_k<<<gblk(nEH), 256, 0, stream>>>(src, dst, EL, ER, MX, HH);
    edge_sum_k<<<gblk(nEH), 256, 0, stream>>>(src, dst, EL, ER, MX, SM, HH);
    edge_scatter_k<<<gblk(nEH * 32), 256, 0, stream>>>(src, dst, EL, ER, MX, SM, FEAT, RST, HH, DD);
    elu_k<<<gblk(nFeat), 256, 0, stream>>>(RST, H0, nFeat);

    // ================= layer 1 (identity residual) =================
    cvt_f32_bf16_k<<<gblk(nFeat), 256, 0, stream>>>(H0, ABF, nFeat);
    gemm256(ABF, W1p, FEAT);
    att_lr_k<<<gblk(nNH * 32), 256, 0, stream>>>(FEAT, al1, ar1, EL, ER, NN, HH, DD);
    init_ms_k<<<gblk(nNH), 256, 0, stream>>>(MX, SM, nNH);
    copy_f32_k<<<gblk(nFeat), 256, 0, stream>>>(H0, RST, nFeat);          // rst = residual
    edge_max_k<<<gblk(nEH), 256, 0, stream>>>(src, dst, EL, ER, MX, HH);
    edge_sum_k<<<gblk(nEH), 256, 0, stream>>>(src, dst, EL, ER, MX, SM, HH);
    edge_scatter_k<<<gblk(nEH * 32), 256, 0, stream>>>(src, dst, EL, ER, MX, SM, FEAT, RST, HH, DD);
    elu_k<<<gblk(nFeat), 256, 0, stream>>>(RST, HOUT, nFeat);             // h -> d_out region

    // ================= output layer (linear residual, mean heads) =================
    cvt_f32_bf16_k<<<gblk(nFeat), 256, 0, stream>>>(HOUT, ABF, nFeat);
    gemm160(ABF, Wop,  FEAT);
    gemm160(ABF, rWop, RST);                                              // rst = res_o
    att_lr_k<<<gblk(nNH * 32), 256, 0, stream>>>(FEAT, alo, aro, EL, ER, NN, HH, OUTC);
    init_ms_k<<<gblk(nNH), 256, 0, stream>>>(MX, SM, nNH);
    edge_max_k<<<gblk(nEH), 256, 0, stream>>>(src, dst, EL, ER, MX, HH);
    edge_sum_k<<<gblk(nEH), 256, 0, stream>>>(src, dst, EL, ER, MX, SM, HH);
    edge_scatter_k<<<gblk(nEH * 32), 256, 0, stream>>>(src, dst, EL, ER, MX, SM, FEAT, RST, HH, OUTC);
    mean_heads_k<<<gblk((long)NN * OUTC), 256, 0, stream>>>(RST, LOGITS, NN, HH, OUTC);
}
